// PNA_9131100472081
// MI455X (gfx1250) — matrix operand order repaired and checked by reading, not compiled, not run on hardware
//
#include <hip/hip_runtime.h>
#include <hip/hip_bf16.h>

#define NN 100000
#define NE 1600000
#define NG 64

typedef __attribute__((ext_vector_type(16))) __bf16 v16bf;
typedef __attribute__((ext_vector_type(8)))  float  v8f;

// ---------- helpers ----------
__device__ inline unsigned short f2bf(float f) {
    unsigned u = __float_as_uint(f);
    u += 0x7FFFu + ((u >> 16) & 1u);
    return (unsigned short)(u >> 16);
}
// monotone float <-> uint order encoding (exact atomic min/max via atomicMin/Max(unsigned))
__device__ inline unsigned ford(float f) {
    unsigned u = __float_as_uint(f);
    return (u & 0x80000000u) ? ~u : (u | 0x80000000u);
}
__device__ inline float ford_inv(unsigned u) {
    unsigned v = (u & 0x80000000u) ? (u & 0x7FFFFFFFu) : ~u;
    return __uint_as_float(v);
}

__device__ inline v8f wmma_bf16(v16bf a, v16bf b, v8f c) {
    return __builtin_amdgcn_wmma_f32_16x16x32_bf16(false, a, false, b, (short)0, c, false, false);
}

// A-fragment (16x32, M x K) directly from bf16 row-major: two b128 loads, no conversion.
// lane<16: M=lane, K elems {kb..kb+7, kb+16..kb+23} with kb=0 ; lane>=16: kb=8
__device__ inline v16bf load_a_frag(const unsigned short* __restrict__ arow, int ksBase, int lane) {
    int khalf = (lane & 16) ? 8 : 0;
    union { v16bf v; uint4 q[2]; } r;
    r.q[0] = *(const uint4*)(arow + ksBase + khalf);
    r.q[1] = *(const uint4*)(arow + ksBase + 16 + khalf);
    return r.v;
}

// B-fragment from LDS in pre-swizzled frag order: frag (ksIdx*4+nt) = 32 lanes x 16 bf16
// contiguous -> two ds_load_b128 per fragment.
__device__ inline v16bf load_b_frag(const unsigned short* __restrict__ sW, int ksIdx, int nt, int lane) {
    const uint4* p = (const uint4*)(sW + ((ksIdx * 4 + nt) * 512) + lane * 16);
    union { v16bf v; uint4 q[2]; } r;
    r.q[0] = p[0]; r.q[1] = p[1];
    return r.v;
}

// ---------- generic WMMA GEMM: C[N x 64] (op)= A[N x 64*KSEG](bf16) @ W[64*KSEG x 64](bf16) ----------
// bias (if used) is a pre-padded 64-entry vector -> unconditional, hoistable loads.
template <int KSEG, bool HAS_BIAS, bool HAS_SCALE, bool ACCUM, bool STOREBF>
__global__ void k_mm(const unsigned short* __restrict__ A, const unsigned short* __restrict__ W,
                     const float* __restrict__ bias, const float* __restrict__ scale,
                     float* __restrict__ C, unsigned short* __restrict__ Cbf, int rowTiles) {
    constexpr int K = 64 * KSEG;
    __shared__ __align__(16) unsigned short sW[K * 64];
    for (int i = threadIdx.x; i < K * 8; i += blockDim.x)
        ((uint4*)sW)[i] = ((const uint4*)W)[i];
    __syncthreads();

    int lane = threadIdx.x & 31;
    int wave = __builtin_amdgcn_readfirstlane(threadIdx.x >> 5);   // wave-uniform -> SGPR
    int col0 = lane & 15;
    int mofs = (lane & 16) ? 8 : 0;

    // bias per lane: 4 values (one per n-tile), loop-invariant
    float bsv[4] = {0.f, 0.f, 0.f, 0.f};
    if constexpr (HAS_BIAS) {
#pragma unroll
        for (int nt = 0; nt < 4; ++nt) bsv[nt] = bias[nt * 16 + col0];
    }

    // K=64: the 8 B-fragments are loop-invariant -> keep in registers, no LDS in the loop
    v16bf breg[KSEG == 1 ? 8 : 1];
    if constexpr (KSEG == 1) {
#pragma unroll
        for (int ks = 0; ks < 2; ++ks)
#pragma unroll
            for (int nt = 0; nt < 4; ++nt) breg[ks * 4 + nt] = load_b_frag(sW, ks, nt, lane);
    }

    for (int rt = blockIdx.x * 8 + wave; rt < rowTiles; rt += gridDim.x * 8) {
        v8f acc[4];
#pragma unroll
        for (int nt = 0; nt < 4; ++nt)
#pragma unroll
            for (int r = 0; r < 8; ++r) acc[nt][r] = 0.f;

        const unsigned short* arow = A + (size_t)(rt * 16 + (lane & 15)) * K;
#pragma unroll
        for (int ks = 0; ks < 2 * KSEG; ++ks) {
            v16bf a = load_a_frag(arow, ks * 32, lane);
#pragma unroll
            for (int nt = 0; nt < 4; ++nt) {
                v16bf b;
                if constexpr (KSEG == 1) b = breg[ks * 4 + nt];
                else                     b = load_b_frag(sW, ks, nt, lane);
                acc[nt] = wmma_bf16(a, b, acc[nt]);
            }
        }
        // epilogue: C/D layout — VGPR r: lanes 0-15 -> M=r, lanes 16-31 -> M=8+r; N=lane%16.
        // One base pointer per tile; all 32 elements at constant offsets r*64 + nt*16.
        size_t tbase = (size_t)(rt * 16 + mofs) * 64 + col0;
        float* cp = C + tbase;
        unsigned short* cbp = nullptr;
        if constexpr (STOREBF) cbp = Cbf + tbase;
        const float* sp = nullptr;
        if constexpr (HAS_SCALE) sp = scale + (rt * 16 + mofs);
#pragma unroll
        for (int r = 0; r < 8; ++r) {
            float sc = 1.f;
            if constexpr (HAS_SCALE) sc = sp[r];
#pragma unroll
            for (int nt = 0; nt < 4; ++nt) {
                int o = r * 64 + nt * 16;           // compile-time constant offset
                float v = acc[nt][r];
                if constexpr (HAS_SCALE) v *= sc;
                if constexpr (HAS_BIAS) v += bsv[nt];
                if constexpr (ACCUM) v += cp[o];
                if constexpr (STOREBF) cbp[o] = f2bf(v);   // final consumer is bf16 GEMM
                else                   cp[o] = v;
            }
        }
    }
}

// ---------- weight packing: src (ld=50) rows [row0..row0+nseg*50) -> bf16 in per-lane
// fragment order: frag f = ksIdx*4+nt (ksIdx in [0,2*nseg)), layout [f][lane][e] ----------
__global__ void k_pack(const float* __restrict__ src, int row0, int nseg, unsigned short* __restrict__ dst) {
    int idx = blockIdx.x * blockDim.x + threadIdx.x;
    if (idx >= nseg * 4096) return;
    int frag = idx >> 9;           // 512 elems per fragment
    int rem  = idx & 511;
    int lane = rem >> 4, e = rem & 15;
    int ks = frag >> 2, nt = frag & 3;
    int Kr = ks * 32 + ((lane & 16) ? 8 : 0) + (e & 7) + ((e & 8) ? 16 : 0);   // padded K row
    int Nc = nt * 16 + (lane & 15);                  // padded N col
    int s = Kr >> 6, rr = Kr & 63;
    float v = 0.f;
    if (rr < 50 && Nc < 50) v = src[(size_t)(row0 + s * 50 + rr) * 50 + Nc];
    dst[idx] = f2bf(v);
}

// pad the 9 bias vectors (pre/post/lin x 3 layers) from 50 to 64 entries, zero-filled
__global__ void k_padb(const float* __restrict__ pre_b, const float* __restrict__ post_b,
                       const float* __restrict__ lin_b, float* __restrict__ b64) {
    int idx = threadIdx.x;              // 0..575
    if (idx >= 576) return;
    int l = idx / 192, r = idx % 192, which = r / 64, f = r % 64;
    const float* s = (which == 0) ? pre_b : (which == 1) ? post_b : lin_b;
    b64[(l * 3 + which) * 64 + f] = (f < 50) ? s[l * 50 + f] : 0.f;
}

// ---------- small utility kernels ----------
__global__ void k_zero(float* p, int n) {
    int i = blockIdx.x * blockDim.x + threadIdx.x;
    if (i < n) p[i] = 0.f;
}

__global__ void k_f2bf(const float* __restrict__ in, unsigned short* __restrict__ out, int n) {
    int i = blockIdx.x * blockDim.x + threadIdx.x;
    if (i < n) out[i] = f2bf(in[i]);
}

__global__ void k_deg(const int* __restrict__ dst, float* __restrict__ deg_raw) {
    int e = blockIdx.x * blockDim.x + threadIdx.x;
    if (e < NE) atomicAdd(&deg_raw[dst[e]], 1.f);
}

__global__ void k_logsum(const float* __restrict__ deg, float* __restrict__ out) {
    __shared__ float s[256];
    int i = blockIdx.x * 256 + threadIdx.x;
    s[threadIdx.x] = (i < NN) ? log1pf(deg[i]) : 0.f;
    __syncthreads();
    for (int st = 128; st > 0; st >>= 1) {
        if (threadIdx.x < st) s[threadIdx.x] += s[threadIdx.x + st];
        __syncthreads();
    }
    if (threadIdx.x == 0) atomicAdd(out, s[0]);
}

__global__ void k_degfin(const float* __restrict__ deg_raw, const float* __restrict__ logsum,
                         float* __restrict__ dinv, float* __restrict__ deg_c,
                         float* __restrict__ amp, float* __restrict__ att) {
    int i = blockIdx.x * blockDim.x + threadIdx.x;
    if (i >= NN) return;
    float dr = deg_raw[i];
    dinv[i] = rsqrtf(dr + 1.f);                 // degree with self-loop >= 1
    float dc = fmaxf(dr, 1.f);
    deg_c[i] = dc;
    float al = logsum[0] * (1.f / NN);
    float lg = log1pf(dc);
    amp[i] = lg / al;
    att[i] = al / lg;
}

// GCN: hs = (x@Wg)*dinv ; H init = hs*dinv (self loop) + b
__global__ void k_gcn_node(const float* __restrict__ x, const float* __restrict__ Wg,
                           const float* __restrict__ bg, const float* __restrict__ dinv,
                           float* __restrict__ hs, float* __restrict__ H) {
    int idx = blockIdx.x * blockDim.x + threadIdx.x;
    int i = idx >> 6, f = idx & 63;
    if (i >= NN) return;
    float v = 0.f;
    if (f < 50) v = x[(size_t)i * 2] * Wg[f] + x[(size_t)i * 2 + 1] * Wg[50 + f];
    float di = dinv[i];
    hs[idx] = v * di;
    H[idx]  = v * di * di + ((f < 50) ? bg[f] : 0.f);
}

__global__ void k_gcn_edge(const int* __restrict__ src, const int* __restrict__ dst,
                           const float* __restrict__ dinv, const float* __restrict__ hs,
                           float* __restrict__ H) {
    int idx = blockIdx.x * blockDim.x + threadIdx.x;
    int e = idx >> 6, f = idx & 63;
    if (e >= NE || f >= 50) return;
    int s = src[e], d = dst[e];
    atomicAdd(&H[(size_t)d * 64 + f], hs[(size_t)s * 64 + f] * dinv[d]);
}

// agg (f32 atomic scratch) layout: [0:64)=sum, [64:128)=min(enc), [128:192)=max(enc), [192:256)=sumsq
__global__ void k_agg_init(float* __restrict__ agg) {
    int idx = blockIdx.x * blockDim.x + threadIdx.x;
    int seg = (idx >> 6) & 3;
    unsigned v = 0;
    if (seg == 1) v = 0xFF800000u;      // ford(+inf)
    else if (seg == 2) v = 0x007FFFFFu; // ford(-inf)
    ((unsigned*)agg)[idx] = v;
}

__global__ void k_pna_edge(const int* __restrict__ src, const int* __restrict__ dst,
                           const float* __restrict__ Ap, const float* __restrict__ Bp,
                           float* __restrict__ agg) {
    int idx = blockIdx.x * blockDim.x + threadIdx.x;
    int e = idx >> 6, f = idx & 63;
    if (e >= NE || f >= 50) return;
    int s = src[e], d = dst[e];
    float h = Ap[(size_t)d * 64 + f] + Bp[(size_t)s * 64 + f];
    float* a = agg + (size_t)d * 256;
    atomicAdd(a + f, h);
    atomicAdd(a + 192 + f, h * h);
    atomicMin((unsigned*)(a + 64) + f, ford(h));
    atomicMax((unsigned*)(a + 128) + f, ford(h));
}

// finalize aggregation straight into bf16 (mean | mn | mx | std), padded with zeros
__global__ void k_agg_fin(const float* __restrict__ agg, const float* __restrict__ deg_raw,
                          const float* __restrict__ deg_c, unsigned short* __restrict__ aggbf) {
    int idx = blockIdx.x * blockDim.x + threadIdx.x;
    int i = idx >> 6, f = idx & 63;
    if (i >= NN) return;
    const float* a = agg + (size_t)i * 256;
    unsigned short* o = aggbf + (size_t)i * 256;
    if (f < 50) {
        float dc = deg_c[i];
        float mean = a[f] / dc;
        float msq  = a[192 + f] / dc;
        float var  = fmaxf(msq - mean * mean, 0.f);
        float sd   = sqrtf(var + 1e-5f);
        bool he = deg_raw[i] > 0.f;
        float mn = he ? ford_inv(__float_as_uint(a[64 + f])) : 0.f;
        float mx = he ? ford_inv(__float_as_uint(a[128 + f])) : 0.f;
        o[f] = f2bf(mean); o[64 + f] = f2bf(mn); o[128 + f] = f2bf(mx); o[192 + f] = f2bf(sd);
    } else {
        o[f] = 0; o[64 + f] = 0; o[128 + f] = 0; o[192 + f] = 0;
    }
}

__global__ void k_bn_part(const float* __restrict__ H, float* __restrict__ stat) {
    __shared__ float ssum[256], ssq[256];
    int f = threadIdx.x & 63;
    int sub = threadIdx.x >> 6;
    float s = 0.f, q = 0.f;
    for (int i = blockIdx.x * 4 + sub; i < NN; i += gridDim.x * 4) {
        float v = H[(size_t)i * 64 + f];
        s += v; q += v * v;
    }
    ssum[threadIdx.x] = s; ssq[threadIdx.x] = q;
    __syncthreads();
    if (sub == 0) {
        s = ssum[f] + ssum[64 + f] + ssum[128 + f] + ssum[192 + f];
        q = ssq[f]  + ssq[64 + f]  + ssq[128 + f]  + ssq[192 + f];
        atomicAdd(&stat[f], s);
        atomicAdd(&stat[64 + f], q);
    }
}

__global__ void k_bn_apply(float* __restrict__ H, unsigned short* __restrict__ Hbf,
                           const float* __restrict__ stat,
                           const float* __restrict__ gamma, const float* __restrict__ beta) {
    int idx = blockIdx.x * blockDim.x + threadIdx.x;
    int i = idx >> 6, f = idx & 63;
    if (i >= NN) return;
    float v = 0.f;
    if (f < 50) {
        const float invN = 1.f / NN;
        float mu  = stat[f] * invN;
        float var = fmaxf(stat[64 + f] * invN - mu * mu, 0.f);
        v = (H[idx] - mu) * rsqrtf(var + 1e-5f) * gamma[f] + beta[f];
        v = fmaxf(v, 0.f);
    }
    H[idx]   = v;
    Hbf[idx] = f2bf(v);
}

__global__ void k_pool(const float* __restrict__ H, const int* __restrict__ batch,
                       float* __restrict__ g) {
    int idx = blockIdx.x * blockDim.x + threadIdx.x;
    int i = idx >> 6, f = idx & 63;
    if (i >= NN || f >= 50) return;
    atomicAdd(&g[(size_t)batch[i] * 64 + f], H[idx]);
}

__global__ void k_mlp(const float* __restrict__ g, const float* __restrict__ W1,
                      const float* __restrict__ b1, const float* __restrict__ W2,
                      const float* __restrict__ b2, float* __restrict__ out) {
    int gi = threadIdx.x;
    if (gi >= NG) return;
    float acc = b2[0];
    for (int j = 0; j < 25; ++j) {
        float s = b1[j];
        for (int f = 0; f < 50; ++f) s += g[(size_t)gi * 64 + f] * W1[f * 25 + j];
        acc += fmaxf(s, 0.f) * W2[j];
    }
    out[gi] = acc;
}

// ---------- host orchestration ----------
extern "C" void kernel_launch(void* const* d_in, const int* in_sizes, int n_in,
                              void* d_out, int out_size, void* d_ws, size_t ws_size,
                              hipStream_t stream) {
    (void)in_sizes; (void)n_in; (void)out_size; (void)ws_size;
    const float* x      = (const float*)d_in[0];
    const int*   ei     = (const int*)d_in[1];
    const int*   batch  = (const int*)d_in[2];
    const float* W_gcn  = (const float*)d_in[3];
    const float* b_gcn  = (const float*)d_in[4];
    const float* pre_W  = (const float*)d_in[5];
    const float* pre_b  = (const float*)d_in[6];
    const float* post_W = (const float*)d_in[7];
    const float* post_b = (const float*)d_in[8];
    const float* lin_W  = (const float*)d_in[9];
    const float* lin_b  = (const float*)d_in[10];
    const float* bn_g   = (const float*)d_in[11];
    const float* bn_b   = (const float*)d_in[12];
    const float* mW1    = (const float*)d_in[13];
    const float* mb1    = (const float*)d_in[14];
    const float* mW2    = (const float*)d_in[15];
    const float* mb2    = (const float*)d_in[16];
    float* out = (float*)d_out;
    const int* src = ei;
    const int* dst = ei + NE;

    // workspace layout (floats first, then bf16 areas); ~310 MB total
    const size_t NF = (size_t)NN * 64;
    float* H    = (float*)d_ws;
    float* hs   = H + NF;
    float* Ap   = H + 2 * NF;
    float* Bp   = H + 3 * NF;
    float* OUT  = H + 4 * NF;
    float* agg  = H + 5 * NF;          // N x 256 (atomic scratch)
    float* scal = H + 9 * NF;
    float* deg_raw = scal;
    float* dinv    = scal + NN;
    float* deg_c   = scal + 2 * (size_t)NN;
    float* amp     = scal + 3 * (size_t)NN;
    float* att     = scal + 4 * (size_t)NN;
    float* logsum  = scal + 5 * (size_t)NN;          // 64
    float* bnstat  = logsum + 64;                    // 128
    float* g       = bnstat + 128;                   // 64*64
    float* bias64  = g + 4096;                       // 9*64 padded biases
    unsigned short* wbase = (unsigned short*)(bias64 + 576);   // 16B-aligned; 3*65536 ushorts
    const size_t LW = 65536;
    unsigned short* Hbf   = wbase + 3 * LW;          // N*64 bf16
    unsigned short* OUTbf = Hbf + NF;                // N*64 bf16
    unsigned short* aggbf = OUTbf + NF;              // N*256 bf16

    const int T = 256;
    const int gNode   = (NN * 64) / T;       // 25000
    const int gEdge   = (NE * 64) / T;       // 400000
    const int gAgg    = NN;                  // N*256/256
    const int rowTiles = NN / 16;            // 6250
    const int gMM = 782;                     // 8 waves/block -> 6256 tiles

    // degrees
    k_zero<<<(NN + T - 1) / T, T, 0, stream>>>(deg_raw, NN);
    k_zero<<<1, 64, 0, stream>>>(logsum, 64);
    k_deg<<<(NE + T - 1) / T, T, 0, stream>>>(dst, deg_raw);
    k_logsum<<<(NN + 255) / 256, 256, 0, stream>>>(deg_raw, logsum);
    k_degfin<<<(NN + T - 1) / T, T, 0, stream>>>(deg_raw, logsum, dinv, deg_c, amp, att);

    // GCN layer
    k_gcn_node<<<gNode, T, 0, stream>>>(x, W_gcn, b_gcn, dinv, hs, H);
    k_gcn_edge<<<gEdge, T, 0, stream>>>(src, dst, dinv, hs, H);
    k_f2bf<<<gNode, T, 0, stream>>>(H, Hbf, NN * 64);

    // pack weights (padded, fragment-swizzled bf16) + pad biases
    k_padb<<<1, 576, 0, stream>>>(pre_b, post_b, lin_b, bias64);
    for (int l = 0; l < 3; ++l) {
        unsigned short* wl = wbase + l * LW;
        k_pack<<<16, T, 0, stream>>>(pre_W + (size_t)l * 5000, 0, 1, wl);                 // Wp_a
        k_pack<<<16, T, 0, stream>>>(pre_W + (size_t)l * 5000, 50, 1, wl + 4096);         // Wp_b
        k_pack<<<16, T, 0, stream>>>(post_W + (size_t)l * 32500, 0, 1, wl + 8192);        // Wq_x
        k_pack<<<16, T, 0, stream>>>(lin_W + (size_t)l * 2500, 0, 1, wl + 12288);         // Wl
        k_pack<<<64, T, 0, stream>>>(post_W + (size_t)l * 32500, 50, 4, wl + 16384);      // Wq1
        k_pack<<<64, T, 0, stream>>>(post_W + (size_t)l * 32500, 250, 4, wl + 32768);     // Wq2
        k_pack<<<64, T, 0, stream>>>(post_W + (size_t)l * 32500, 450, 4, wl + 49152);     // Wq3
    }

    // PNA layers
    for (int l = 0; l < 3; ++l) {
        unsigned short* wl = wbase + l * LW;
        const float* bpre  = bias64 + (l * 3 + 0) * 64;
        const float* bpost = bias64 + (l * 3 + 1) * 64;
        const float* blin  = bias64 + (l * 3 + 2) * 64;
        // edge-MLP factorized to node matmuls
        k_mm<1, true,  false, false, false><<<gMM, T, 0, stream>>>(Hbf, wl,        bpre,    nullptr, Ap,  nullptr, rowTiles);
        k_mm<1, false, false, false, false><<<gMM, T, 0, stream>>>(Hbf, wl + 4096, nullptr, nullptr, Bp,  nullptr, rowTiles);
        // aggregation
        k_agg_init<<<gAgg, T, 0, stream>>>(agg);
        k_pna_edge<<<gEdge, T, 0, stream>>>(src, dst, Ap, Bp, agg);
        k_agg_fin<<<gNode, T, 0, stream>>>(agg, deg_raw, deg_c, aggbf);
        // post: OUT = H@Wq_x + bq + agg@Wq1 + amp*(agg@Wq2) + att*(agg@Wq3) ; last one emits bf16
        k_mm<1, true,  false, false, false><<<gMM, T, 0, stream>>>(Hbf,   wl + 8192,  bpost,   nullptr, OUT, nullptr, rowTiles);
        k_mm<4, false, false, true,  false><<<gMM, T, 0, stream>>>(aggbf, wl + 16384, nullptr, nullptr, OUT, nullptr, rowTiles);
        k_mm<4, false, true,  true,  false><<<gMM, T, 0, stream>>>(aggbf, wl + 32768, nullptr, amp,     OUT, nullptr, rowTiles);
        k_mm<4, false, true,  true,  true ><<<gMM, T, 0, stream>>>(aggbf, wl + 49152, nullptr, att,     OUT, OUTbf,   rowTiles);
        // linear
        k_mm<1, true,  false, false, false><<<gMM, T, 0, stream>>>(OUTbf, wl + 12288, blin,    nullptr, H,   nullptr, rowTiles);
        // batch norm + relu (also refresh bf16 copy of H for next layer's GEMMs)
        k_zero<<<1, 128, 0, stream>>>(bnstat, 128);
        k_bn_part<<<256, 256, 0, stream>>>(H, bnstat);
        k_bn_apply<<<gNode, T, 0, stream>>>(H, Hbf, bnstat, bn_g + l * 50, bn_b + l * 50);
    }

    // readout
    k_zero<<<16, T, 0, stream>>>(g, 4096);
    k_pool<<<gNode, T, 0, stream>>>(H, batch, g);
    k_mlp<<<1, 64, 0, stream>>>(g, mW1, mb1, mW2, mb2, out);
}
